// SequentialVAE_27315992003195
// MI455X (gfx1250) — matrix operand order repaired and checked by reading, not compiled, not run on hardware
//
#include <hip/hip_runtime.h>
#include <hip/hip_bf16.h>

typedef __attribute__((ext_vector_type(16))) _Float16 vh16;
typedef __attribute__((ext_vector_type(8)))  _Float16 vh8;
typedef __attribute__((ext_vector_type(8)))  float    vf8;

#define TM1 99
#define NB  128

// ---------------- workspace layout (float offsets) ----------------
static const size_t OFF_AFT   = 0;          // [99][128][16]
static const size_t OFF_OFT   = 202752;     // [99][128][32]
static const size_t OFF_HALL  = 608256;     // [99][128][64]
static const size_t OFF_MUM   = 1419264;    // [99][128][8]
static const size_t OFF_LVM   = 1520640;
static const size_t OFF_MUMT  = 1622016;
static const size_t OFF_LVMT  = 1723392;
static const size_t OFF_DECF  = 1824768;    // [99][128][32]
static const size_t OFF_DECH0 = 2230272;    // [99][128][64]
static const size_t OFF_DPART = 3041280;    // [99]
static const size_t OFF_KPART = 3041379;    // [99]
static const size_t OFF_HALF  = 3041536;    // f16 region base (32B aligned)
// half-element offsets inside f16 region
static const size_t HO_ENCWI = 0;           // [192][64] (K padded 48->64)
static const size_t HO_ENCWH = 12288;       // [192][64]
static const size_t HO_DECWI = 24576;       // [192][64]
static const size_t HO_DECWH = 36864;       // [192][64]
static const size_t HO_ADW1  = 49152;       // [64][64]
static const size_t HO_ADW2  = 53248;       // [7][64]

struct Params { const float* w[48]; };
// dict order: 0 ae_w1 1 ae_b1 2 ae_w2 3 ae_b2 | 4 oe_w1 5 oe_b1 6 oe_w2 7 oe_b2
// 8 enc_Wi 9 enc_Wh 10 enc_bi 11 enc_bh | 12..15 mu | 16..19 lv | 20..23 el
// 24..27 mut | 28..31 lvt | 32..35 llt | 36 dec_Wi 37 dec_Wh 38 dec_bi 39 dec_bh
// 40..43 ll | 44 ad_w1 45 ad_b1 46 ad_w2 47 ad_b2

// ---------------- WMMA helpers ----------------
__device__ __forceinline__ vf8 wmma16(vh16 a, vh16 b, vf8 c) {
  return __builtin_amdgcn_wmma_f32_16x16x32_f16(false, a, false, b, (short)0, c, false, false);
}
__device__ __forceinline__ vf8 zero8() { vf8 z; for (int i = 0; i < 8; ++i) z[i] = 0.f; return z; }

// A-frag (16x32 f16): lane l holds row M = mtile*16 + (l&15);
// halves i -> K = ks*32 + 16*hi + i (16 contiguous halves)
__device__ __forceinline__ vh16 load_a_frag(const _Float16* base, int mtile, int ks, int lane) {
  int lo = lane & 15, hi = lane >> 4;
  const _Float16* p = base + ((size_t)(mtile * 16 + lo)) * 64 + ks * 32 + hi * 16;
  vh8 x0 = *(const vh8*)(p);
  vh8 x1 = *(const vh8*)(p + 8);
  vh16 a;
  for (int i = 0; i < 8; ++i) { a[i] = x0[i]; a[i + 8] = x1[i]; }
  return a;
}
// B-frag (32x16 f16) from row-major W[N][64]: lane l holds col N = ntile*16 + (l&15),
// halves i -> K = ks*32 + 16*hi + i (16 contiguous halves)
__device__ __forceinline__ vh16 load_b_frag(const _Float16* W, int ntile, int ks, int lane) {
  int lo = lane & 15, hi = lane >> 4;
  return *(const vh16*)(W + ((size_t)(ntile * 16 + lo)) * 64 + ks * 32 + hi * 16);
}
__device__ __forceinline__ float sigmf(float x) { return 1.f / (1.f + __expf(-x)); }

// ---------------- prep: embeddings, transposes, f16 weights ----------------
__global__ void __launch_bounds__(256) k_prep(const float* __restrict__ obs,
                                              const float* __restrict__ act, Params P,
                                              float* afT, float* ofT,
                                              _Float16* eWi, _Float16* eWh, _Float16* dWi,
                                              _Float16* dWh, _Float16* aw1, _Float16* aw2) {
  int id = blockIdx.x * 256 + threadIdx.x;
  if (id < 12672) {                       // action MLP 1->16->16, transposed store
    int t = id / NB, b = id % NB;
    float a = act[b * 100 + t];
    float h[16];
    for (int i = 0; i < 16; ++i) h[i] = fmaxf(P.w[0][i] * a + P.w[1][i], 0.f);
    for (int i = 0; i < 16; ++i) {
      float s = P.w[3][i];
      for (int j = 0; j < 16; ++j) s += P.w[2][i * 16 + j] * h[j];
      afT[(size_t)id * 16 + i] = s;
    }
  } else if (id < 25344) {                // obs MLP 11->32->32, transposed store
    int q = id - 12672; int t = q / NB, b = q % NB;
    const float* o = obs + ((size_t)b * 100 + t) * 11;
    float h[32];
    for (int i = 0; i < 32; ++i) {
      float s = P.w[5][i];
      for (int k = 0; k < 11; ++k) s += P.w[4][i * 11 + k] * o[k];
      h[i] = fmaxf(s, 0.f);
    }
    for (int i = 0; i < 32; ++i) {
      float s = P.w[7][i];
      for (int j = 0; j < 32; ++j) s += P.w[6][i * 32 + j] * h[j];
      ofT[(size_t)q * 32 + i] = s;
    }
  } else if (id < 37632) { int q = id - 25344; int n = q / 64, k = q % 64;
    eWi[q] = (_Float16)(k < 48 ? P.w[8][n * 48 + k] : 0.f);
  } else if (id < 49920) { int q = id - 37632; eWh[q] = (_Float16)P.w[9][q];
  } else if (id < 62208) { int q = id - 49920; dWi[q] = (_Float16)P.w[36][q];
  } else if (id < 74496) { int q = id - 62208; dWh[q] = (_Float16)P.w[37][q];
  } else if (id < 78592) { int q = id - 74496; aw1[q] = (_Float16)P.w[44][q];
  } else if (id < 79040) { int q = id - 78592; aw2[q] = (_Float16)P.w[46][q]; }
}

// ---------------- encoder GRU: 1 block, 8 waves, Mtile-per-wave ----------------
__global__ void __launch_bounds__(256) k_encoder(const float* __restrict__ afT,
                                                 const float* __restrict__ ofT,
                                                 const _Float16* __restrict__ eWi,
                                                 const _Float16* __restrict__ eWh,
                                                 const float* __restrict__ bi,
                                                 const float* __restrict__ bh,
                                                 float* __restrict__ Hall) {
  __shared__ __attribute__((aligned(32))) _Float16 sWi[192 * 64];
  __shared__ __attribute__((aligned(32))) _Float16 sWh[192 * 64];
  __shared__ __attribute__((aligned(32))) _Float16 sA[128 * 64];
  int tid = threadIdx.x, lane = tid & 31, w = tid >> 5;
  int lo = lane & 15, hi = lane >> 4;
  for (int i = tid; i < 192 * 64; i += 256) { sWi[i] = eWi[i]; sWh[i] = eWh[i]; }
  float bi_[4][3], bh_[4][3];
  for (int c = 0; c < 4; ++c)
    for (int g = 0; g < 3; ++g) { int col = g * 64 + c * 16 + lo; bi_[c][g] = bi[col]; bh_[c][g] = bh[col]; }
  vf8 hfrag[4];
  for (int c = 0; c < 4; ++c) hfrag[c] = zero8();
  __syncthreads();
  for (int t = 0; t < TM1; ++t) {
    for (int i = tid; i < 128 * 64; i += 256) {        // stage X = [af|of|0]
      int row = i >> 6, col = i & 63;
      float v = (col < 16) ? afT[((size_t)t * 128 + row) * 16 + col]
              : (col < 48) ? ofT[((size_t)t * 128 + row) * 32 + (col - 16)] : 0.f;
      sA[i] = (_Float16)v;
    }
    __syncthreads();
    vh16 a0 = load_a_frag(sA, w, 0, lane), a1 = load_a_frag(sA, w, 1, lane);
    vf8 gi[4][3];
    for (int c = 0; c < 4; ++c)
      for (int g = 0; g < 3; ++g) {
        vf8 acc = zero8();
        acc = wmma16(a0, load_b_frag(sWi, c + 4 * g, 0, lane), acc);
        acc = wmma16(a1, load_b_frag(sWi, c + 4 * g, 1, lane), acc);
        gi[c][g] = acc;
      }
    __syncthreads();
    for (int c = 0; c < 4; ++c)                        // stage H (from regs)
      for (int r = 0; r < 8; ++r)
        sA[(size_t)(w * 16 + r + 8 * hi) * 64 + c * 16 + lo] = (_Float16)hfrag[c][r];
    __syncthreads();
    vh16 ha0 = load_a_frag(sA, w, 0, lane), ha1 = load_a_frag(sA, w, 1, lane);
    for (int c = 0; c < 4; ++c) {
      vf8 gh[3];
      for (int g = 0; g < 3; ++g) {
        vf8 acc = zero8();
        acc = wmma16(ha0, load_b_frag(sWh, c + 4 * g, 0, lane), acc);
        acc = wmma16(ha1, load_b_frag(sWh, c + 4 * g, 1, lane), acc);
        gh[g] = acc;
      }
      for (int r = 0; r < 8; ++r) {
        float rg = sigmf(gi[c][0][r] + bi_[c][0] + gh[0][r] + bh_[c][0]);
        float zg = sigmf(gi[c][1][r] + bi_[c][1] + gh[1][r] + bh_[c][1]);
        float ng = tanhf(gi[c][2][r] + bi_[c][2] + rg * (gh[2][r] + bh_[c][2]));
        float hn = (1.f - zg) * ng + zg * hfrag[c][r];
        hfrag[c][r] = hn;
        Hall[((size_t)t * 128 + (w * 16 + r + 8 * hi)) * 64 + c * 16 + lo] = hn;
      }
    }
    __syncthreads();
  }
}

// ---------------- per-step latent heads: one block per (t,b) ----------------
__device__ __forceinline__ float dotn(const float* a, const float* w, int n) {
  float s = 0.f;
  for (int i = 0; i < n; ++i) s += a[i] * w[i];
  return s;
}
__global__ void __launch_bounds__(64) k_heads(const float* __restrict__ Hall,
                                              const float* __restrict__ eps_m,
                                              const float* __restrict__ eps_mt, Params P,
                                              float* mu_m, float* lv_m, float* mu_mt,
                                              float* lv_mt, float* dec_f, float* dec_h0,
                                              float* out) {
  int t = blockIdx.x / NB, b = blockIdx.x % NB, tid = threadIdx.x;
  size_t tb = (size_t)t * NB + b;
  __shared__ float sh[64], st[64], sfeat[64], sm[8], smt[8], smu[8], scat[40];
  sh[tid] = Hall[tb * 64 + tid];
  __syncthreads();
  st[tid] = fmaxf(dotn(sh, P.w[12] + tid * 64, 64) + P.w[13][tid], 0.f);  __syncthreads();
  if (tid < 8) { float v = dotn(st, P.w[14] + tid * 64, 64) + P.w[15][tid]; smu[tid] = v; mu_m[tb * 8 + tid] = v; }
  __syncthreads();
  st[tid] = fmaxf(dotn(sh, P.w[16] + tid * 64, 64) + P.w[17][tid], 0.f);  __syncthreads();
  if (tid < 8) {
    float v = dotn(st, P.w[18] + tid * 64, 64) + P.w[19][tid]; lv_m[tb * 8 + tid] = v;
    float m = smu[tid] + eps_m[tb * 8 + tid] * __expf(0.5f * v);
    sm[tid] = m; out[2 + ((size_t)b * TM1 + t) * 8 + tid] = m;            // perm_latents
  }
  __syncthreads();
  st[tid] = fmaxf(dotn(sh, P.w[20] + tid * 64, 64) + P.w[21][tid], 0.f);  __syncthreads();
  sfeat[tid] = dotn(st, P.w[22] + tid * 64, 64) + P.w[23][tid];           __syncthreads();
  st[tid] = fmaxf(dotn(sfeat, P.w[24] + tid * 64, 64) + P.w[25][tid], 0.f); __syncthreads();
  if (tid < 8) { float v = dotn(st, P.w[26] + tid * 64, 64) + P.w[27][tid]; smu[tid] = v; mu_mt[tb * 8 + tid] = v; }
  __syncthreads();
  st[tid] = fmaxf(dotn(sfeat, P.w[28] + tid * 64, 64) + P.w[29][tid], 0.f); __syncthreads();
  if (tid < 8) {
    float v = dotn(st, P.w[30] + tid * 64, 64) + P.w[31][tid]; lv_mt[tb * 8 + tid] = v;
    float mt = smu[tid] + eps_mt[tb * 8 + tid] * __expf(0.5f * v);
    smt[tid] = mt; out[2 + 101376 + ((size_t)b * TM1 + t) * 8 + tid] = mt; // var_latents
  }
  __syncthreads();
  if (tid < 32) st[tid] = fmaxf(dotn(sm, P.w[40] + tid * 8, 8) + P.w[41][tid], 0.f);
  __syncthreads();
  if (tid < 32) { float v = dotn(st, P.w[42] + tid * 32, 32) + P.w[43][tid]; scat[tid] = v; dec_f[tb * 32 + tid] = v; }
  __syncthreads();
  if (tid >= 32 && tid < 40) scat[tid] = smt[tid - 32];
  __syncthreads();
  st[tid] = fmaxf(dotn(scat, P.w[32] + tid * 40, 40) + P.w[33][tid], 0.f); __syncthreads();
  dec_h0[tb * 64 + tid] = dotn(st, P.w[34] + tid * 64, 64) + P.w[35][tid];
}

// ---------------- KL: one block per t, deterministic reduce ----------------
__global__ void __launch_bounds__(128) k_kl(const float* mu_m, const float* lv_m,
                                            const float* mu_mt, const float* lv_mt,
                                            float* kpart) {
  int t = blockIdx.x, b = threadIdx.x;
  size_t tb = (size_t)t * NB + b, pb = (size_t)(t - 1) * NB + b;
  float s = 0.f;
  for (int k = 0; k < 16; ++k) {
    float cl = (k < 8) ? lv_m[tb * 8 + k] : lv_mt[tb * 8 + k - 8];
    float cm = (k < 8) ? mu_m[tb * 8 + k] : mu_mt[tb * 8 + k - 8];
    float pm = 0.f, pl = 0.f;
    if (t > 0) {  // prior swaps mu<->logvar, exactly as the reference
      pm = (k < 8) ? lv_m[pb * 8 + k] : lv_mt[pb * 8 + k - 8];
      pl = (k < 8) ? mu_m[pb * 8 + k] : mu_mt[pb * 8 + k - 8];
    }
    float d = cm - pm;
    s += pl - cl + (__expf(cl) + d * d) * __expf(-pl) - 1.f;
  }
  __shared__ float red[128];
  red[b] = s;
  __syncthreads();
  if (b == 0) { float a = 0.f; for (int i = 0; i < 128; ++i) a += red[i]; kpart[t] = a; }
}

// ---------------- decoder: fused GRU + logits + NLL, one block per t ----------------
__global__ void __launch_bounds__(256) k_decoder(const float* __restrict__ ofT,
                                                 const float* __restrict__ dec_f,
                                                 const float* __restrict__ dec_h0,
                                                 const _Float16* __restrict__ dWi,
                                                 const _Float16* __restrict__ dWh,
                                                 const _Float16* __restrict__ aw1,
                                                 const _Float16* __restrict__ aw2,
                                                 const float* __restrict__ bi,
                                                 const float* __restrict__ bh,
                                                 const float* __restrict__ ab1,
                                                 const float* __restrict__ ab2,
                                                 const float* __restrict__ act,
                                                 float* __restrict__ dpart) {
  __shared__ __attribute__((aligned(32))) _Float16 sWi[192 * 64];
  __shared__ __attribute__((aligned(32))) _Float16 sWh[192 * 64];
  __shared__ __attribute__((aligned(32))) _Float16 sA[128 * 64];
  int t = blockIdx.x;
  int tid = threadIdx.x, lane = tid & 31, w = tid >> 5;
  int lo = lane & 15, hi = lane >> 4;
  for (int i = tid; i < 192 * 64; i += 256) { sWi[i] = dWi[i]; sWh[i] = dWh[i]; }
  float bi_[4][3], bh_[4][3], ab1_[4];
  for (int c = 0; c < 4; ++c) {
    for (int g = 0; g < 3; ++g) { int col = g * 64 + c * 16 + lo; bi_[c][g] = bi[col]; bh_[c][g] = bh[col]; }
    ab1_[c] = ab1[c * 16 + lo];
  }
  vf8 hfrag[4];
  for (int c = 0; c < 4; ++c)
    for (int r = 0; r < 8; ++r)
      hfrag[c][r] = dec_h0[((size_t)t * 128 + (w * 16 + r + 8 * hi)) * 64 + c * 16 + lo];
  float accum = 0.f;
  __syncthreads();
  for (int j = t; j < TM1; ++j) {
    for (int i = tid; i < 128 * 64; i += 256) {        // stage X = [of_j | dec_f_t]
      int row = i >> 6, col = i & 63;
      float v = (col < 32) ? ofT[((size_t)j * 128 + row) * 32 + col]
                           : dec_f[((size_t)t * 128 + row) * 32 + (col - 32)];
      sA[i] = (_Float16)v;
    }
    if (j + 1 < TM1) __builtin_prefetch(ofT + ((size_t)(j + 1) * 128 + tid) * 32, 0, 1);
    __syncthreads();
    vh16 a0 = load_a_frag(sA, w, 0, lane), a1 = load_a_frag(sA, w, 1, lane);
    vf8 gi[4][3];
    for (int c = 0; c < 4; ++c)
      for (int g = 0; g < 3; ++g) {
        vf8 acc = zero8();
        acc = wmma16(a0, load_b_frag(sWi, c + 4 * g, 0, lane), acc);
        acc = wmma16(a1, load_b_frag(sWi, c + 4 * g, 1, lane), acc);
        gi[c][g] = acc;
      }
    __syncthreads();
    for (int c = 0; c < 4; ++c)
      for (int r = 0; r < 8; ++r)
        sA[(size_t)(w * 16 + r + 8 * hi) * 64 + c * 16 + lo] = (_Float16)hfrag[c][r];
    __syncthreads();
    vh16 ha0 = load_a_frag(sA, w, 0, lane), ha1 = load_a_frag(sA, w, 1, lane);
    for (int c = 0; c < 4; ++c) {
      vf8 gh[3];
      for (int g = 0; g < 3; ++g) {
        vf8 acc = zero8();
        acc = wmma16(ha0, load_b_frag(sWh, c + 4 * g, 0, lane), acc);
        acc = wmma16(ha1, load_b_frag(sWh, c + 4 * g, 1, lane), acc);
        gh[g] = acc;
      }
      for (int r = 0; r < 8; ++r) {
        float rg = sigmf(gi[c][0][r] + bi_[c][0] + gh[0][r] + bh_[c][0]);
        float zg = sigmf(gi[c][1][r] + bi_[c][1] + gh[1][r] + bh_[c][1]);
        float ng = tanhf(gi[c][2][r] + bi_[c][2] + rg * (gh[2][r] + bh_[c][2]));
        hfrag[c][r] = (1.f - zg) * ng + zg * hfrag[c][r];
      }
    }
    __syncthreads();
    for (int c = 0; c < 4; ++c)                        // stage relu(h) for logits MLP
      for (int r = 0; r < 8; ++r)
        sA[(size_t)(w * 16 + r + 8 * hi) * 64 + c * 16 + lo] = (_Float16)fmaxf(hfrag[c][r], 0.f);
    __syncthreads();
    vh16 ra0 = load_a_frag(sA, w, 0, lane), ra1 = load_a_frag(sA, w, 1, lane);
    vf8 a1acc[4];
    for (int c = 0; c < 4; ++c) {
      vf8 acc = zero8();
      acc = wmma16(ra0, load_b_frag(aw1, c, 0, lane), acc);
      acc = wmma16(ra1, load_b_frag(aw1, c, 1, lane), acc);
      a1acc[c] = acc;
    }
    __syncthreads();
    for (int c = 0; c < 4; ++c)                        // stage relu(a1 + b1)
      for (int r = 0; r < 8; ++r)
        sA[(size_t)(w * 16 + r + 8 * hi) * 64 + c * 16 + lo] =
            (_Float16)fmaxf(a1acc[c][r] + ab1_[c], 0.f);
    __syncthreads();
    if (tid < 128) {                                   // logits 64->7, log-softmax, NLL
      const _Float16* rowp = sA + (size_t)tid * 64;
      float lg[7];
      for (int k = 0; k < 7; ++k) {
        float s = ab2[k];
        const _Float16* wk = aw2 + k * 64;
        for (int d = 0; d < 64; ++d) s += (float)rowp[d] * (float)wk[d];
        lg[k] = s;
      }
      float mx = lg[0];
      for (int k = 1; k < 7; ++k) mx = fmaxf(mx, lg[k]);
      float se = 0.f;
      for (int k = 0; k < 7; ++k) se += __expf(lg[k] - mx);
      float lse = mx + __logf(se);
      int tgt = (int)act[tid * 100 + (j + 1)];
      accum += lse - lg[tgt];
    }
    __syncthreads();
  }
  float* red = (float*)sA;                             // reuse LDS as f32 scratch
  if (tid < 128) red[tid] = accum;
  __syncthreads();
  if (tid == 0) { float s = 0.f; for (int i = 0; i < 128; ++i) s += red[i]; dpart[t] = s; }
}

__global__ void k_final(const float* dpart, const float* kpart, float* out) {
  if (threadIdx.x == 0 && blockIdx.x == 0) {
    float s = 0.f, k = 0.f;
    for (int i = 0; i < TM1; ++i) { s += dpart[i]; k += kpart[i]; }
    out[0] = s / 128.f;
    out[1] = 0.5f * k / 128.f;
  }
}

extern "C" void kernel_launch(void* const* d_in, const int* in_sizes, int n_in,
                              void* d_out, int out_size, void* d_ws, size_t ws_size,
                              hipStream_t stream) {
  const float* obs    = (const float*)d_in[0];
  const float* act    = (const float*)d_in[1];
  const float* eps_m  = (const float*)d_in[2];
  const float* eps_mt = (const float*)d_in[3];
  Params P;
  for (int i = 0; i < 48; ++i) P.w[i] = (const float*)d_in[4 + i];
  float* out = (float*)d_out;
  float* ws  = (float*)d_ws;
  float* afT = ws + OFF_AFT;   float* ofT = ws + OFF_OFT;   float* Hall = ws + OFF_HALL;
  float* mum = ws + OFF_MUM;   float* lvm = ws + OFF_LVM;
  float* mumt = ws + OFF_MUMT; float* lvmt = ws + OFF_LVMT;
  float* decf = ws + OFF_DECF; float* dech0 = ws + OFF_DECH0;
  float* dpart = ws + OFF_DPART; float* kpart = ws + OFF_KPART;
  _Float16* hbase = (_Float16*)(ws + OFF_HALF);
  _Float16* eWi = hbase + HO_ENCWI; _Float16* eWh = hbase + HO_ENCWH;
  _Float16* dWi = hbase + HO_DECWI; _Float16* dWh = hbase + HO_DECWH;
  _Float16* aw1 = hbase + HO_ADW1;  _Float16* aw2 = hbase + HO_ADW2;

  k_prep<<<(79040 + 255) / 256, 256, 0, stream>>>(obs, act, P, afT, ofT,
                                                  eWi, eWh, dWi, dWh, aw1, aw2);
  k_encoder<<<1, 256, 0, stream>>>(afT, ofT, eWi, eWh, P.w[10], P.w[11], Hall);
  k_heads<<<TM1 * NB, 64, 0, stream>>>(Hall, eps_m, eps_mt, P, mum, lvm, mumt, lvmt,
                                       decf, dech0, out);
  k_kl<<<TM1, 128, 0, stream>>>(mum, lvm, mumt, lvmt, kpart);
  k_decoder<<<TM1, 256, 0, stream>>>(ofT, decf, dech0, dWi, dWh, aw1, aw2,
                                     P.w[38], P.w[39], P.w[45], P.w[47], act, dpart);
  k_final<<<1, 32, 0, stream>>>(dpart, kpart, out);
}
